// SpatialAttention_46179488366786
// MI455X (gfx1250) — compile-verified, matrix operand order checked by reading, not run on hardware
//
#include <hip/hip_runtime.h>

#define B_   8
#define T_   64
#define N_   128
#define P2_  256
#define KS_  64
#define BT_  (B_ * T_)

typedef __bf16 bf16;
typedef __bf16 v16bf __attribute__((ext_vector_type(16)));
typedef __bf16 v8bf  __attribute__((ext_vector_type(8)));
typedef float  v8f   __attribute__((ext_vector_type(8)));
typedef float  v4f   __attribute__((ext_vector_type(4)));
typedef int    v4i   __attribute__((ext_vector_type(4)));

#define AS1 __attribute__((address_space(1)))
#define AS3 __attribute__((address_space(3)))

// gfx1250 async global->LDS path (ASYNCcnt), guarded so host pass and
// builtin-less toolchains fall back to plain vector copies.
#if defined(__AMDGCN__) && __has_builtin(__builtin_amdgcn_global_load_async_to_lds_b128)
#define ASYNC_LDS 1
#else
#define ASYNC_LDS 0
#endif

// Copy 16 bytes (8 bf16) global -> LDS.
__device__ __forceinline__ void copy16_g2l(const bf16* g, bf16* l) {
#if ASYNC_LDS
  __builtin_amdgcn_global_load_async_to_lds_b128(
      (AS1 v4i*)(bf16*)g, (AS3 v4i*)l, 0, 0);
#else
  *(v8bf*)l = *(const v8bf*)g;
#endif
}

__device__ __forceinline__ void async_wait_all() {
#if ASYNC_LDS
#if __has_builtin(__builtin_amdgcn_s_wait_asynccnt)
  __builtin_amdgcn_s_wait_asynccnt(0);
#else
  asm volatile("s_wait_asynccnt 0" ::: "memory");
#endif
#endif
}

union FragU { v16bf v; v8bf h[2]; };

// Load a 16x32 bf16 WMMA operand fragment from row-major [row][K] memory.
// Per CDNA5 16-bit A/B layout: lanes 0-15 -> K{k0..k0+7} (elems 0-7) and
// K{k0+16..k0+23} (elems 8-15); lanes 16-31 -> K{k0+8..} and K{k0+24..}.
__device__ __forceinline__ v16bf load_frag(const bf16* __restrict__ base,
                                           int stride, int row0, int k0, int lane) {
  int r  = row0 + (lane & 15);
  int kb = k0 + ((lane >> 4) << 3);
  const bf16* p = base + r * stride + kb;
  FragU f;
  f.h[0] = *(const v8bf*)(p);
  f.h[1] = *(const v8bf*)(p + 16);
  return f.v;
}

__device__ __forceinline__ v8f wmma_bf16(v16bf a, v16bf b, v8f c) {
  return __builtin_amdgcn_wmma_f32_16x16x32_bf16(false, a, false, b,
                                                 (short)0, c, false, false);
}

__device__ __forceinline__ v8f zero8() {
  v8f z;
#pragma unroll
  for (int i = 0; i < 8; ++i) z[i] = 0.f;
  return z;
}

// D-tile layout: element (M = v + 8*(lane>=16), N = lane&15) in VGPR v.
// Transposed store: out[n][m], 8 consecutive m per lane -> one v8bf / v8f store.
__device__ __forceinline__ void store_tile_T_bf16(bf16* __restrict__ out, int stride,
                                                  int m0, int n0, v8f acc, int lane) {
  int n  = n0 + (lane & 15);
  int mb = m0 + ((lane >> 4) << 3);
  v8bf o;
#pragma unroll
  for (int i = 0; i < 8; ++i) o[i] = (bf16)acc[i];
  *(v8bf*)(out + n * stride + mb) = o;
}

__device__ __forceinline__ void store_tile_T_f32(float* __restrict__ out, int stride,
                                                 int m0, int n0, v8f acc, int lane,
                                                 float scale) {
  int n  = n0 + (lane & 15);
  int mb = m0 + ((lane >> 4) << 3);
  v8f o;
#pragma unroll
  for (int i = 0; i < 8; ++i) o[i] = acc[i] * scale;
  *(v8f*)(out + n * stride + mb) = o;
}

// ---------------------------------------------------------------------------
// Kernel 0: convert + transpose weights to bf16.
// wqt/wkt: [KS][P2], wvt: [P2][P2] (i.e. W^T row-major, K=p contiguous).
// ---------------------------------------------------------------------------
__global__ __launch_bounds__(256) void prep_w_kernel(
    const float* __restrict__ wq, const float* __restrict__ wk,
    const float* __restrict__ wv,
    bf16* __restrict__ wqt, bf16* __restrict__ wkt, bf16* __restrict__ wvt) {
  int idx = blockIdx.x * 256 + threadIdx.x;          // 0 .. 65535
  if (idx < KS_ * P2_) {
    int k = idx >> 8, p = idx & 255;
    wqt[idx] = (bf16)wq[p * KS_ + k];
    wkt[idx] = (bf16)wk[p * KS_ + k];
  }
  if (idx < P2_ * P2_) {
    int k = idx >> 8, p = idx & 255;
    wvt[idx] = (bf16)wv[p * P2_ + k];
  }
}

// ---------------------------------------------------------------------------
// Kernel 1: fused QKV projection per (b,t) tile. x[b,t] (128x256 f32) is
// staged once into LDS as bf16; three GEMMs against L2-resident weights.
//   qT = WQT(64x256) @ xT   -> transposed store -> qws [i][k]   (stride 64)
//   kT = WKT        @ xT    -> transposed store -> kws [j][k]
//   v  = x @ WV  (B op = WVT[q][p]) tile(M=j,N=q) -> T-store -> vtws [q][j]
// ---------------------------------------------------------------------------
__global__ __launch_bounds__(256) void qkv_kernel(
    const float* __restrict__ x,
    const bf16* __restrict__ wqt, const bf16* __restrict__ wkt,
    const bf16* __restrict__ wvt,
    bf16* __restrict__ qws, bf16* __restrict__ kws, bf16* __restrict__ vtws) {
  __shared__ bf16 xs[N_ * P2_];                       // 64 KB
  int bt  = blockIdx.x;
  int tid = threadIdx.x;
  int lane = tid & 31;
  int wave = tid >> 5;                                // 0..7

  // Prefetch L2-resident transposed weights (global_prefetch_b8).
  __builtin_prefetch(wvt + tid * 256, 0, 1);
  __builtin_prefetch(wqt + (tid & 63) * 256, 0, 1);
  __builtin_prefetch(wkt + (tid & 63) * 256, 0, 1);

  const float* xg = x + (size_t)bt * (N_ * P2_);
#pragma unroll 4
  for (int i = tid * 4; i < N_ * P2_; i += 256 * 4) {
    v4f xv = *(const v4f*)(xg + i);
#pragma unroll
    for (int c = 0; c < 4; ++c) xs[i + c] = (bf16)xv[c];
  }
  __syncthreads();

  bf16* qo = qws  + (size_t)bt * (N_ * KS_);
  bf16* ko = kws  + (size_t)bt * (N_ * KS_);
  bf16* vo = vtws + (size_t)bt * (P2_ * N_);

  // Q^T and K^T: 2 * (4 m-tiles x 8 n-tiles) = 64 tiles, K = 256 (8 steps).
  for (int t = wave; t < 64; t += 8) {
    int which = t >> 5;
    int tt = t & 31;
    int mt = tt >> 3;                                 // k_out tile (0..3)
    int nt = tt & 7;                                  // i tile (0..7)
    const bf16* wt = which ? wkt : wqt;
    bf16* o = which ? ko : qo;
    v8f acc = zero8();
#pragma unroll
    for (int ks = 0; ks < 8; ++ks) {
      v16bf a = load_frag(wt, P2_, mt * 16, ks * 32, lane);
      v16bf b = load_frag(xs, P2_, nt * 16, ks * 32, lane);
      acc = wmma_bf16(a, b, acc);
    }
    store_tile_T_bf16(o, KS_, mt * 16, nt * 16, acc, lane);
  }

  // V: 8 m-tiles (j) x 16 n-tiles (q) = 128 tiles, K = 256 (8 steps).
  for (int t = wave; t < 128; t += 8) {
    int mt = t & 7;                                   // j tile
    int nt = t >> 3;                                  // q tile (0..15)
    v8f acc = zero8();
#pragma unroll
    for (int ks = 0; ks < 8; ++ks) {
      v16bf a = load_frag(xs,  P2_, mt * 16, ks * 32, lane);
      v16bf b = load_frag(wvt, P2_, nt * 16, ks * 32, lane);
      acc = wmma_bf16(a, b, acc);
    }
    store_tile_T_bf16(vo, N_, mt * 16, nt * 16, acc, lane);   // -> vtws[q][j]
  }
}

// ---------------------------------------------------------------------------
// Kernel 2: scores^T = k @ q^T per (b,t). q,k tiles (16 KB each) staged in
// LDS via async global->LDS copies; transposed D store yields scores[i][j]
// f32 with 1/sqrt(64) folded in.
// ---------------------------------------------------------------------------
__global__ __launch_bounds__(256) void scores_kernel(
    const bf16* __restrict__ qws, const bf16* __restrict__ kws,
    float* __restrict__ sws) {
  __shared__ bf16 qs[N_ * KS_];                       // 16 KB
  __shared__ bf16 ksh[N_ * KS_];                      // 16 KB
  int bt = blockIdx.x;
  int tid = threadIdx.x;
  int lane = tid & 31;
  int wave = tid >> 5;
  const bf16* qg = qws + (size_t)bt * (N_ * KS_);
  const bf16* kg = kws + (size_t)bt * (N_ * KS_);
#pragma unroll
  for (int i = tid * 8; i < N_ * KS_; i += 256 * 8) {
    copy16_g2l(qg + i, qs + i);
    copy16_g2l(kg + i, ksh + i);
  }
  async_wait_all();
  __syncthreads();

  float* s = sws + (size_t)bt * (N_ * N_);
  for (int t = wave; t < 64; t += 8) {
    int mt = t >> 3;                                  // j tile
    int nt = t & 7;                                   // i tile
    v8f acc = zero8();
#pragma unroll
    for (int ks = 0; ks < 2; ++ks) {
      v16bf a = load_frag(ksh, KS_, mt * 16, ks * 32, lane);
      v16bf b = load_frag(qs,  KS_, nt * 16, ks * 32, lane);
      acc = wmma_bf16(a, b, acc);
    }
    store_tile_T_f32(s, N_, mt * 16, nt * 16, acc, lane, 0.125f);
  }
}

// ---------------------------------------------------------------------------
// Kernel 3: softmax over the T axis (axis=1). One thread per (b,i,j);
// 3 strided passes over T=64, all hits in the 192 MB L2. attn stored bf16.
// ---------------------------------------------------------------------------
__global__ __launch_bounds__(256) void softmax_t_kernel(
    const float* __restrict__ sws, bf16* __restrict__ aws) {
  int idx = blockIdx.x * 256 + threadIdx.x;           // 0 .. B*N*N-1
  int b  = idx >> 14;
  int ij = idx & 16383;
  const float* sp = sws + (size_t)b * T_ * (N_ * N_) + ij;
  bf16* ap = aws + (size_t)b * T_ * (N_ * N_) + ij;
  float m = -3.402823466e38f;
#pragma unroll 4
  for (int t = 0; t < T_; ++t) m = fmaxf(m, sp[t * (N_ * N_)]);
  float sum = 0.f;
#pragma unroll 4
  for (int t = 0; t < T_; ++t) sum += __expf(sp[t * (N_ * N_)] - m);
  float inv = 1.f / sum;
#pragma unroll 4
  for (int t = 0; t < T_; ++t)
    ap[t * (N_ * N_)] = (bf16)(__expf(sp[t * (N_ * N_)] - m) * inv);
}

// ---------------------------------------------------------------------------
// Kernel 4: out^T = v^T @ attn^T per (b,t). attn (32 KB) + v^T (64 KB)
// staged in LDS via async copies (96 KB, 3 blocks/WGP on the 320 KB LDS):
//   A = v^T [q][j] (M=q, 16 tiles), B = attn [i][j] (N=i, 8 tiles), K=128.
// Transposed D store -> out[i][q] f32 contiguous.
// ---------------------------------------------------------------------------
__global__ __launch_bounds__(256) void out_kernel(
    const bf16* __restrict__ aws, const bf16* __restrict__ vtws,
    float* __restrict__ out) {
  __shared__ bf16 ash[N_ * N_];                       // 32 KB
  __shared__ bf16 vsh[P2_ * N_];                      // 64 KB
  int bt = blockIdx.x;
  int tid = threadIdx.x;
  int lane = tid & 31;
  int wave = tid >> 5;
  const bf16* ag = aws  + (size_t)bt * (N_ * N_);
  const bf16* vg = vtws + (size_t)bt * (P2_ * N_);
#pragma unroll
  for (int i = tid * 8; i < N_ * N_; i += 256 * 8)
    copy16_g2l(ag + i, ash + i);
#pragma unroll
  for (int i = tid * 8; i < P2_ * N_; i += 256 * 8)
    copy16_g2l(vg + i, vsh + i);
  async_wait_all();
  __syncthreads();

  float* o = out + (size_t)bt * (N_ * P2_);
  for (int t = wave; t < 128; t += 8) {
    int mt = t >> 3;                                  // q tile (0..15)
    int nt = t & 7;                                   // i tile
    v8f acc = zero8();
#pragma unroll
    for (int ks = 0; ks < 4; ++ks) {
      v16bf af = load_frag(vsh, N_, mt * 16, ks * 32, lane);
      v16bf bf = load_frag(ash, N_, nt * 16, ks * 32, lane);
      acc = wmma_bf16(af, bf, acc);
    }
    store_tile_T_f32(o, P2_, mt * 16, nt * 16, acc, lane, 1.f);
  }
}

extern "C" void kernel_launch(void* const* d_in, const int* in_sizes, int n_in,
                              void* d_out, int out_size, void* d_ws, size_t ws_size,
                              hipStream_t stream) {
  const float* x  = (const float*)d_in[0];
  const float* wq = (const float*)d_in[1];
  const float* wk = (const float*)d_in[2];
  const float* wv = (const float*)d_in[3];
  float* out = (float*)d_out;

  char* ws = (char*)d_ws;
  size_t off = 0;
  bf16* wqt = (bf16*)(ws + off); off += (size_t)KS_ * P2_ * 2;          // 32 KB
  bf16* wkt = (bf16*)(ws + off); off += (size_t)KS_ * P2_ * 2;          // 32 KB
  bf16* wvt = (bf16*)(ws + off); off += (size_t)P2_ * P2_ * 2;          // 128 KB
  bf16* qws = (bf16*)(ws + off); off += (size_t)BT_ * N_ * KS_ * 2;     // 8 MB
  bf16* kws = (bf16*)(ws + off); off += (size_t)BT_ * N_ * KS_ * 2;     // 8 MB
  bf16* vtw = (bf16*)(ws + off); off += (size_t)BT_ * P2_ * N_ * 2;     // 32 MB
  float* sws = (float*)(ws + off); off += (size_t)BT_ * N_ * N_ * 4;    // 32 MB
  bf16* aws = (bf16*)(ws + off); off += (size_t)BT_ * N_ * N_ * 2;      // 16 MB

  prep_w_kernel<<<(P2_ * P2_) / 256, 256, 0, stream>>>(wq, wk, wv, wqt, wkt, wvt);
  qkv_kernel<<<BT_, 256, 0, stream>>>(x, wqt, wkt, wvt, qws, kws, vtw);
  scores_kernel<<<BT_, 256, 0, stream>>>(qws, kws, sws);
  softmax_t_kernel<<<(B_ * N_ * N_) / 256, 256, 0, stream>>>(sws, aws);
  out_kernel<<<BT_, 256, 0, stream>>>(aws, vtw, out);
}
